// FuseDR_Net_35442070126656
// MI455X (gfx1250) — matrix operand order repaired and checked by reading, not compiled, not run on hardware
//
#include <hip/hip_runtime.h>
#include <hip/hip_bf16.h>
#include <math.h>

typedef __attribute__((ext_vector_type(16))) _Float16 v16h;
typedef __attribute__((ext_vector_type(8)))  float    v8f;

#define BATCH 32
#define SEQL  512
#define DMODEL 256
#define NHEADS 8
#define DHEAD  32
#define DINNER 512
#define DSTATE 16
#define DTRANK 16
#define DFF    512
#define PREDL  96
#define ROWS   (BATCH*SEQL)       // 16384
#define CALEN  259                // wavelet coeff length
#define CATOT  (BATCH*DMODEL*CALEN)

// ---- gfx1250 async global->LDS path (ASYNCcnt), guarded for toolchain ----
#if defined(__gfx1250__) && __has_builtin(__builtin_amdgcn_global_load_async_to_lds_b128) && __has_builtin(__builtin_amdgcn_s_wait_asynccnt)
#define HAS_ASYNC_LDS 1
#else
#define HAS_ASYNC_LDS 0
#endif

typedef int v4i_ __attribute__((vector_size(16)));
#if HAS_ASYNC_LDS
typedef __attribute__((address_space(1))) v4i_* g128_t;   // global (AS1) 16B vector ptr
typedef __attribute__((address_space(3))) v4i_* l128_t;   // LDS    (AS3) 16B vector ptr
#endif

__device__ __forceinline__ void async_cp16(const float* g, float* l) {
#if HAS_ASYNC_LDS
  __builtin_amdgcn_global_load_async_to_lds_b128(
      (g128_t)(size_t)g,
      (l128_t)(unsigned)(size_t)l,
      0, 0);
#else
  (void)g; (void)l;
#endif
}
__device__ __forceinline__ void async_wait0() {
#if HAS_ASYNC_LDS
  __builtin_amdgcn_s_wait_asynccnt(0);
#endif
}

// ---------------- wavelet filter constants ----------------
__constant__ float c_lo[8]  = {-0.010597401784997278f, 0.032883011666982945f, 0.030841381835986965f,
                               -0.18703481171888114f, -0.02798376941698385f, 0.6308807679295904f,
                                0.7148465705525415f,   0.23037781330885523f};
__constant__ float c_hi[8]  = {-0.23037781330885523f,  0.7148465705525415f, -0.6308807679295904f,
                               -0.02798376941698385f,  0.18703481171888114f, 0.030841381835986965f,
                               -0.032883011666982945f, -0.010597401784997278f};
__constant__ float c_lor[8] = { 0.23037781330885523f,  0.7148465705525415f,  0.6308807679295904f,
                               -0.02798376941698385f, -0.18703481171888114f, 0.030841381835986965f,
                                0.032883011666982945f, -0.010597401784997278f};
__constant__ float c_hir[8] = {-0.010597401784997278f, -0.032883011666982945f, 0.030841381835986965f,
                                0.18703481171888114f, -0.02798376941698385f, -0.6308807679295904f,
                                0.7148465705525415f,  -0.23037781330885523f};

// ---------------- activations ----------------
template<int ACT> __device__ __forceinline__ float act_fn(float x) {
  if (ACT == 1) return x / (1.f + __expf(-x));                               // silu
  if (ACT == 2) {                                                            // gelu(tanh)
    float x3 = x*x*x;
    return 0.5f*x*(1.f + tanhf(0.7978845608028654f*(x + 0.044715f*x3)));
  }
  if (ACT == 3) return 1.f / (1.f + __expf(-x));                             // sigmoid
  if (ACT == 4) return (x > 20.f) ? x : log1pf(__expf(x));                   // softplus
  return x;
}

// =====================================================================
// WMMA GEMM: Y[r,n] = act( sum_k X[r,k] * W[n,k] + bias[n] ) + Res[r,n]
// 128 threads (4 waves), tile 64(M) x 64(N), K-step 32.
// Double-buffered LDS panels staged with async global->LDS when possible;
// f32 in LDS, f32->f16 folded into fragment build, f32 accumulate.
// =====================================================================
template<int ACT>
__global__ __launch_bounds__(128)
void wmma_gemm_kernel(const float* __restrict__ X, int ldx,
                      const float* __restrict__ Wm, const float* __restrict__ bias,
                      float* __restrict__ Y, int ldy,
                      const float* __restrict__ Res,
                      int Rrows, int K, int N)
{
  __shared__ float Asf[2][64][32];   // 8 KB x2  (rows x k)
  __shared__ float Wsf[2][64][32];   // 8 KB x2  (cols x k, un-transposed)
  const int tid  = threadIdx.x;
  const int wv   = tid >> 5;
  const int lane = tid & 31;
  const int lo   = lane & 15;
  const int half = lane >> 4;
  const int row0 = blockIdx.y * 64;
  const int col0 = blockIdx.x * 64;

  v8f acc[4];
  acc[0] = v8f{}; acc[1] = v8f{}; acc[2] = v8f{}; acc[3] = v8f{};

  auto stage = [&](int buf, int kk) {
    const bool kfull = (kk + 32 <= K);
#if HAS_ASYNC_LDS
    if (kfull && (row0 + 64 <= Rrows)) {
      #pragma unroll
      for (int c = tid; c < 512; c += 128) {          // 512 x 16B chunks
        int r = c >> 3, q4 = (c & 7) << 2;
        async_cp16(&X[(size_t)(row0 + r)*ldx + kk + q4], &Asf[buf][r][q4]);
      }
    } else
#endif
    {
      for (int i = tid; i < 2048; i += 128) {
        int r = i >> 5, cc = i & 31;
        int gr = row0 + r, gk = kk + cc;
        Asf[buf][r][cc] = (gr < Rrows && gk < K) ? X[(size_t)gr*ldx + gk] : 0.f;
      }
    }
#if HAS_ASYNC_LDS
    if (kfull && (col0 + 64 <= N)) {
      #pragma unroll
      for (int c = tid; c < 512; c += 128) {
        int n = c >> 3, q4 = (c & 7) << 2;
        async_cp16(&Wm[(size_t)(col0 + n)*K + kk + q4], &Wsf[buf][n][q4]);
      }
    } else
#endif
    {
      for (int i = tid; i < 2048; i += 128) {
        int n = i >> 5, k = i & 31;
        int gn = col0 + n, gk = kk + k;
        Wsf[buf][n][k] = (gn < N && gk < K) ? Wm[(size_t)gn*K + gk] : 0.f;
      }
    }
  };

  stage(0, 0);
  async_wait0();
  __syncthreads();

  int buf = 0;
  for (int kk0 = 0; kk0 < K; kk0 += 32) {
    if (kk0 + 32 < K) stage(buf ^ 1, kk0 + 32);      // prefetch next panel (async)

    // A fragment: m = 16*wave + lo ; e<8 -> K=8*half+e ; e>=8 -> K=16+8*half+(e-8)
    v16h a;
    const int m = (wv << 4) | lo;
    #pragma unroll
    for (int e = 0; e < 8; ++e) a[e]     = (_Float16)Asf[buf][m][(half << 3) + e];
    #pragma unroll
    for (int e = 0; e < 8; ++e) a[8 + e] = (_Float16)Asf[buf][m][16 + (half << 3) + e];

    // 4 N-tiles: B fragment n = 16*nt + lo ; e<8 -> K=8*half+e ; e>=8 -> K=16+8*half+(e-8)
    #pragma unroll
    for (int nt = 0; nt < 4; ++nt) {
      v16h bf;
      #pragma unroll
      for (int e = 0; e < 16; ++e) bf[e] = (_Float16)Wsf[buf][(nt << 4) + lo][(half << 3) + (e & 7) + ((e >> 3) << 4)];
      acc[nt] = __builtin_amdgcn_wmma_f32_16x16x32_f16(false, a, false, bf, (short)0, acc[nt], false, false);
    }

    async_wait0();          // next panel's async copies complete
    __syncthreads();        // all waves done reading buf; next panel visible
    buf ^= 1;
  }

  // epilogue: C elem j -> row = 8*half + j, col = lo
  #pragma unroll
  for (int nt = 0; nt < 4; ++nt) {
    int gn = col0 + (nt << 4) + lo;
    if (gn >= N) continue;
    #pragma unroll
    for (int j = 0; j < 8; ++j) {
      int gm = row0 + (wv << 4) + (half << 3) + j;
      if (gm >= Rrows) continue;
      float v = acc[nt][j];
      if (bias) v += bias[gn];
      v = act_fn<ACT>(v);
      if (Res) v += Res[(size_t)gm*ldy + gn];
      Y[(size_t)gm*ldy + gn] = v;
    }
  }
}

// =====================================================================
// Fused causal flash-attention: one wave per (q-tile16, b*h).
// 32-key chunks: 2 S-WMMAs + full 16x32 P fragment + 2 P@V WMMAs.
// Online softmax on WMMA C-fragment layout via 16-lane shfl butterflies.
// =====================================================================
__global__ __launch_bounds__(32)
void attn_kernel(const float* __restrict__ Q, const float* __restrict__ Kb,
                 const float* __restrict__ V, float* __restrict__ O)
{
  __shared__ _Float16 Pl[16][32];
  const int lane = threadIdx.x;
  const int lo   = lane & 15;
  const int half = lane >> 4;
  const int qt   = blockIdx.x;           // 0..31
  const int bh   = blockIdx.y;           // 0..255
  const int b    = bh >> 3, h = bh & 7;
  const int q0   = qt << 4;
  const int cbase = h * DHEAD;
  const float scale = 0.17677669529663687f;   // 1/sqrt(32)

  // Q tile -> A fragment (16 x 32)
  v16h aq;
  {
    const float* qr = Q + ((size_t)(b*SEQL + q0 + lo))*DMODEL + cbase;
    #pragma unroll
    for (int e = 0; e < 8; ++e) aq[e]     = (_Float16)qr[(half << 4) + e];
    #pragma unroll
    for (int e = 0; e < 8; ++e) aq[8 + e] = (_Float16)qr[8 + (half << 4) + e];
  }

  v8f acc0 = {}, acc1 = {};
  float mrun[8], lrun[8];
  #pragma unroll
  for (int j = 0; j < 8; ++j) { mrun[j] = -3.0e38f; lrun[j] = 0.f; }

  const int last = (q0 + 15) >> 5;       // last 32-key chunk index
  for (int c = 0; c <= last; ++c) {
    const int k0 = c << 5;
    const bool dia = (c == last);

    // K^T tiles -> B fragments: keys [k0,k0+15] and [k0+16,k0+31]
    v16h bk0, bk1;
    {
      const float* kr0 = Kb + ((size_t)(b*SEQL + k0 + lo))*DMODEL + cbase + (half << 4);
      const float* kr1 = kr0 + (size_t)16*DMODEL;
      #pragma unroll
      for (int e = 0; e < 16; ++e) { bk0[e] = (_Float16)kr0[e]; bk1[e] = (_Float16)kr1[e]; }
    }
    v8f zc = {};
    v8f s0 = __builtin_amdgcn_wmma_f32_16x16x32_f16(false, aq, false, bk0, (short)0, zc, false, false);
    v8f s1 = __builtin_amdgcn_wmma_f32_16x16x32_f16(false, aq, false, bk1, (short)0, zc, false, false);

    #pragma unroll
    for (int j = 0; j < 8; ++j) {
      int mrow = (half << 3) + j;
      float sv0 = s0[j] * scale;
      float sv1 = s1[j] * scale;
      if (dia) {
        if (k0 + lo      > q0 + mrow) sv0 = -1.0e30f;
        if (k0 + 16 + lo > q0 + mrow) sv1 = -1.0e30f;
      }
      float cm = fmaxf(sv0, sv1);
      cm = fmaxf(cm, __shfl_xor(cm, 1));
      cm = fmaxf(cm, __shfl_xor(cm, 2));
      cm = fmaxf(cm, __shfl_xor(cm, 4));
      cm = fmaxf(cm, __shfl_xor(cm, 8));
      float mnew = fmaxf(mrun[j], cm);
      float p0 = __expf(sv0 - mnew);
      float p1 = __expf(sv1 - mnew);
      float ps = p0 + p1;
      ps += __shfl_xor(ps, 1); ps += __shfl_xor(ps, 2);
      ps += __shfl_xor(ps, 4); ps += __shfl_xor(ps, 8);
      float corr = __expf(mrun[j] - mnew);
      lrun[j] = lrun[j]*corr + ps;
      mrun[j] = mnew;
      acc0[j] *= corr; acc1[j] *= corr;
      Pl[mrow][lo]      = (_Float16)p0;
      Pl[mrow][16 + lo] = (_Float16)p1;
    }
    __syncthreads();

    // P (16x32) -> A fragment (full K)
    v16h ap;
    #pragma unroll
    for (int e = 0; e < 8; ++e) ap[e]     = Pl[lo][(half << 4) + e];
    #pragma unroll
    for (int e = 0; e < 8; ++e) ap[8 + e] = Pl[lo][8 + (half << 4) + e];

    // V tiles -> B fragments: elem e -> key k0 + 16*half + e
    v16h bv0, bv1;
    #pragma unroll
    for (int e = 0; e < 16; ++e) {
      const float* vr = V + ((size_t)(b*SEQL + k0 + (half << 4) + e))*DMODEL + cbase;
      bv0[e] = (_Float16)vr[lo];
      bv1[e] = (_Float16)vr[16 + lo];
    }
    acc0 = __builtin_amdgcn_wmma_f32_16x16x32_f16(false, ap, false, bv0, (short)0, acc0, false, false);
    acc1 = __builtin_amdgcn_wmma_f32_16x16x32_f16(false, ap, false, bv1, (short)0, acc1, false, false);
    __syncthreads();
  }

  #pragma unroll
  for (int j = 0; j < 8; ++j) {
    int m = q0 + (half << 3) + j;
    float inv = 1.f / lrun[j];
    float* orow = O + ((size_t)(b*SEQL + m))*DMODEL + cbase;
    orow[lo]      = acc0[j] * inv;
    orow[16 + lo] = acc1[j] * inv;
  }
}

// =====================================================================
// Data embedding: circular-pad k=3 conv over feature channels + sinusoidal PE
// =====================================================================
__global__ void embed_kernel(const float* __restrict__ xe, const float* __restrict__ cw,
                             float* __restrict__ out, int F, int which)
{
  int idx = blockIdx.x * blockDim.x + threadIdx.x;
  if (idx >= ROWS * DMODEL) return;
  int d = idx & 255;
  int l = (idx >> 8) & 511;
  int b = idx >> 17;
  float acc = 0.f;
  for (int ff = 0; ff < F; ++ff) {
    int ch;
    if (which == 0) ch = (ff == 0) ? 17 : 0;                           // [trend, cycle]
    else            ch = (ff < 16) ? (ff + 1) : ((ff == 16) ? 18 : 0); // [health16, noise, cycle]
    const float* wrow = cw + ((size_t)d*F + ff)*3;
    #pragma unroll
    for (int tt = 0; tt < 3; ++tt) {
      int lm = (l + tt - 1 + SEQL) & 511;
      acc += wrow[tt] * xe[((size_t)(b*SEQL + lm))*19 + ch];
    }
  }
  int de = d & ~1;
  float div = __expf(-(float)de * (9.210340371976184f / 256.f));   // ln(1e4)/256
  float pe = (d & 1) ? __cosf((float)l * div) : __sinf((float)l * div);
  out[idx] = acc + pe;
}

// =====================================================================
// LayerNorm over D=256 (optionally fused residual add): out = LN(src[+add])*g+b
// =====================================================================
__global__ __launch_bounds__(256)
void ln_kernel(const float* __restrict__ src, const float* __restrict__ add,
               const float* __restrict__ g, const float* __restrict__ bb,
               float* __restrict__ out)
{
  __shared__ float red[256];
  const size_t row = blockIdx.x;
  const int c = threadIdx.x;
  float v = src[row*DMODEL + c];
  if (add) v += add[row*DMODEL + c];
  red[c] = v; __syncthreads();
  for (int s = 128; s > 0; s >>= 1) { if (c < s) red[c] += red[c + s]; __syncthreads(); }
  float mean = red[0] * (1.f/256.f);
  __syncthreads();
  float dv = v - mean;
  red[c] = dv*dv; __syncthreads();
  for (int s = 128; s > 0; s >>= 1) { if (c < s) red[c] += red[c + s]; __syncthreads(); }
  float var = red[0] * (1.f/256.f);
  out[row*DMODEL + c] = dv * rsqrtf(var + 1e-5f) * g[c] + bb[c];
}

// depthwise causal conv k=4 over xm = xz[:, :DINNER], + bias, silu
__global__ void dwconv_silu_kernel(const float* __restrict__ xz, const float* __restrict__ cw,
                                   const float* __restrict__ cb, float* __restrict__ xc)
{
  int idx = blockIdx.x * blockDim.x + threadIdx.x;
  if (idx >= ROWS * DINNER) return;
  int c = idx & 511;
  int l = (idx >> 9) & 511;
  int b = idx >> 18;
  float acc = cb[c];
  #pragma unroll
  for (int kk = 0; kk < 4; ++kk) {
    int lm = l - 3 + kk;
    if (lm >= 0) acc += cw[c*4 + kk] * xz[((size_t)(b*SEQL + lm))*(2*DINNER) + c];
  }
  xc[idx] = acc / (1.f + __expf(-acc));   // silu
}

// selective scan: thread = (b, d); state[16] in regs; B/C staged per-t in LDS
__global__ __launch_bounds__(256)
void scan_kernel(const float* __restrict__ xdbl, const float* __restrict__ dtb,
                 const float* __restrict__ xc, const float* __restrict__ xz,
                 const float* __restrict__ A_log, const float* __restrict__ Dp,
                 float* __restrict__ Y)
{
  __shared__ float bc[32];
  const int b = blockIdx.x >> 1;
  const int d = ((blockIdx.x & 1) << 8) + threadIdx.x;   // 0..511
  float a[DSTATE], hst[DSTATE];
  #pragma unroll
  for (int n = 0; n < DSTATE; ++n) { a[n] = -__expf(A_log[d*DSTATE + n]); hst[n] = 0.f; }
  const float dval = Dp[d];
  for (int t = 0; t < SEQL; ++t) {
    const size_t rt = (size_t)(b*SEQL + t);
    if (threadIdx.x < 32) bc[threadIdx.x] = xdbl[rt*48 + DTRANK + threadIdx.x];
    __syncthreads();
    const float dt = dtb[rt*DINNER + d];
    const float xv = xc[rt*DINNER + d];
    const float zv = xz[rt*(2*DINNER) + DINNER + d];
    float acc = 0.f;
    #pragma unroll
    for (int n = 0; n < DSTATE; ++n) {
      hst[n] = __expf(dt * a[n]) * hst[n] + (dt * xv) * bc[n];
      acc += hst[n] * bc[16 + n];
    }
    Y[rt*DINNER + d] = (acc + xv*dval) * (zv / (1.f + __expf(-zv)));
    __syncthreads();
  }
}

// DWT analysis: symmetric pad 6/6, stride-2 correlation with reversed filters
__global__ void dwt_kernel(const float* __restrict__ X, float* __restrict__ cA, float* __restrict__ cD)
{
  int idx = blockIdx.x * blockDim.x + threadIdx.x;
  if (idx >= CATOT) return;
  int i = idx % CALEN;
  int d = (idx / CALEN) & 255;
  int b = idx / (CALEN * DMODEL);
  float av = 0.f, dv = 0.f;
  #pragma unroll
  for (int k = 0; k < 8; ++k) {
    int p = 2*i + k;
    int l = (p < 6) ? (5 - p) : ((p >= 518) ? (1029 - p) : (p - 6));
    float x = X[((size_t)(b*SEQL + l))*DMODEL + d];
    av += c_lor[k] * x;
    dv += c_hir[k] * x;
  }
  cA[idx] = av; cD[idx] = dv;
}

// im2col for causal k=3 conv over coeff length: rows = B*259, K = 256*3
__global__ void im2col_kernel(const float* __restrict__ src, float* __restrict__ dst)
{
  int idx = blockIdx.x * blockDim.x + threadIdx.x;
  if (idx >= BATCH * CALEN * 768) return;
  int col = idx % 768;
  int t = (idx / 768) % CALEN;
  int b = idx / (768 * CALEN);
  int c = col / 3, kk = col % 3;
  int tt = t - 2 + kk;
  dst[idx] = (tt >= 0) ? src[((size_t)(b*DMODEL + c))*CALEN + tt] : 0.f;
}

// IDWT synthesis (lhs_dilation=2, pad (1,1)) summed over LO/HI branches
__global__ void recon_kernel(const float* __restrict__ caf, const float* __restrict__ cdf,
                             float* __restrict__ rect)
{
  int idx = blockIdx.x * blockDim.x + threadIdx.x;
  if (idx >= ROWS * DMODEL) return;
  int d = idx & 255;
  int l = (idx >> 8) & 511;
  int b = idx >> 17;
  float r = 0.f;
  #pragma unroll
  for (int k = 0; k < 8; ++k) {
    int q = l + k - 1;
    if (q >= 0 && !(q & 1)) {
      int i = q >> 1;
      if (i < CALEN) {
        size_t o = ((size_t)(b*CALEN + i))*DMODEL + d;
        r += c_lo[k]*caf[o] + c_hi[k]*cdf[o];
      }
    }
  }
  rect[idx] = r;
}

// comb = t + f * gate(sigmoid already applied)
__global__ void combine_kernel(const float* __restrict__ t, const float* __restrict__ f,
                               const float* __restrict__ g, float* __restrict__ out)
{
  int idx = blockIdx.x * blockDim.x + threadIdx.x;
  if (idx >= ROWS * DMODEL) return;
  out[idx] = t[idx] + f[idx] * g[idx];
}

// final projection, only last PREDL positions
__global__ void proj_kernel(const float* __restrict__ comb, const float* __restrict__ pw,
                            const float* __restrict__ pb, float* __restrict__ out)
{
  int idx = blockIdx.x * blockDim.x + threadIdx.x;
  if (idx >= BATCH * PREDL) return;
  int b = idx / PREDL;
  int i = idx % PREDL;
  int l = SEQL - PREDL + i;
  const float* row = comb + ((size_t)(b*SEQL + l))*DMODEL;
  float acc = pb[0];
  #pragma unroll 8
  for (int d = 0; d < DMODEL; ++d) acc += row[d] * pw[d];
  out[idx] = acc;
}

// ---------------- host-side GEMM dispatcher ----------------
static inline void gemm(hipStream_t s, int act, const float* X, int ldx,
                        const float* Wm, const float* bias, float* Y, int ldy,
                        const float* res, int R, int K, int N)
{
  dim3 g((N + 63) / 64, (R + 63) / 64), b(128);
  switch (act) {
    case 1: wmma_gemm_kernel<1><<<g, b, 0, s>>>(X, ldx, Wm, bias, Y, ldy, res, R, K, N); break;
    case 2: wmma_gemm_kernel<2><<<g, b, 0, s>>>(X, ldx, Wm, bias, Y, ldy, res, R, K, N); break;
    case 3: wmma_gemm_kernel<3><<<g, b, 0, s>>>(X, ldx, Wm, bias, Y, ldy, res, R, K, N); break;
    case 4: wmma_gemm_kernel<4><<<g, b, 0, s>>>(X, ldx, Wm, bias, Y, ldy, res, R, K, N); break;
    default: wmma_gemm_kernel<0><<<g, b, 0, s>>>(X, ldx, Wm, bias, Y, ldy, res, R, K, N); break;
  }
}

extern "C" void kernel_launch(void* const* d_in, const int* in_sizes, int n_in,
                              void* d_out, int out_size, void* d_ws, size_t ws_size,
                              hipStream_t stream)
{
  (void)in_sizes; (void)n_in; (void)out_size; (void)ws_size;
  const float* xenc      = (const float*)d_in[0];
  const float* trend_emb = (const float*)d_in[1];
  const float* fluct_emb = (const float*)d_in[2];
  const float* gate_w    = (const float*)d_in[3];
  const float* gate_b    = (const float*)d_in[4];
  const float* proj_w    = (const float*)d_in[5];
  const float* proj_b    = (const float*)d_in[6];
  #define MB(i,k) ((const float*)d_in[7  + (i)*19 + (k)])
  #define AT(i,k) ((const float*)d_in[45 + (i)*10 + (k)])
  #define WVp(i,k) ((const float*)d_in[65 + (i)*6 + (k)])

  const size_t R = ROWS;
  float* ws   = (float*)d_ws;
  float* t    = ws;                 // R*256
  float* f    = t    + R*256;       // R*256
  float* hbuf = f    + R*256;       // R*256
  float* xz   = hbuf + R*256;       // R*1024 (reused slab)
  float* xcv  = xz   + R*1024;      // R*512  (reused slab)
  float* xdbl = xcv  + R*512;       // R*48
  float* dtb  = xdbl + R*48;        // R*512  (reused slab)
  float* ybuf = dtb  + R*512;       // R*512  (reused slab)

  const int EB = 256;
  embed_kernel<<<(ROWS*DMODEL + EB - 1)/EB, EB, 0, stream>>>(xenc, trend_emb, t, 2, 0);
  embed_kernel<<<(ROWS*DMODEL + EB - 1)/EB, EB, 0, stream>>>(xenc, fluct_emb, f, 18, 1);

  // ---------- Mamba stack on trend stream ----------
  for (int i = 0; i < 2; ++i) {
    ln_kernel<<<(int)R, 256, 0, stream>>>(t, nullptr, MB(i,0), MB(i,1), hbuf);
    gemm(stream, 0, hbuf, 256, MB(i,2), MB(i,3), xz, 1024, nullptr, (int)R, 256, 1024);     // in_proj
    dwconv_silu_kernel<<<(int)((R*512 + 255)/256), 256, 0, stream>>>(xz, MB(i,4), MB(i,5), xcv);
    gemm(stream, 0, xcv, 512, MB(i,6), nullptr, xdbl, 48, nullptr, (int)R, 512, 48);        // x_proj
    gemm(stream, 4, xdbl, 48, MB(i,7), MB(i,8), dtb, 512, nullptr, (int)R, 16, 512);        // dt + softplus
    scan_kernel<<<64, 256, 0, stream>>>(xdbl, dtb, xcv, xz, MB(i,9), MB(i,10), ybuf);       // selective scan
    gemm(stream, 0, ybuf, 512, MB(i,11), MB(i,12), t, 256, t, (int)R, 512, 256);            // out_proj + res
    ln_kernel<<<(int)R, 256, 0, stream>>>(t, nullptr, MB(i,13), MB(i,14), hbuf);
    gemm(stream, 2, hbuf, 256, MB(i,15), MB(i,16), xcv, 512, nullptr, (int)R, 256, 512);    // ff1 + gelu
    gemm(stream, 0, xcv, 512, MB(i,17), MB(i,18), t, 256, t, (int)R, 512, 256);             // ff2 + res
  }

  // ---------- attention + wavelet stack on fluctuation stream ----------
  float* qb = dtb;
  float* kb = dtb + R*256;
  float* vb = ybuf;
  float* ao = ybuf + R*256;
  float* cA   = xz;
  float* cD   = xz + (size_t)CATOT;
  float* caf  = xz + (size_t)2*CATOT;
  float* cdf  = xz + (size_t)3*CATOT;
  float* rect = xz + (size_t)4*CATOT;
  float* colb = xcv;

  for (int i = 0; i < 2; ++i) {
    gemm(stream, 0, f, 256, AT(i,0), AT(i,1), qb, 256, nullptr, (int)R, 256, 256);
    gemm(stream, 0, f, 256, AT(i,2), AT(i,3), kb, 256, nullptr, (int)R, 256, 256);
    gemm(stream, 0, f, 256, AT(i,4), AT(i,5), vb, 256, nullptr, (int)R, 256, 256);
    attn_kernel<<<dim3(SEQL/16, BATCH*NHEADS), 32, 0, stream>>>(qb, kb, vb, ao);
    gemm(stream, 0, ao, 256, AT(i,6), AT(i,7), xz, 256, nullptr, (int)R, 256, 256);         // o_proj
    ln_kernel<<<(int)R, 256, 0, stream>>>(xz, f, AT(i,8), AT(i,9), f);                      // LN(x+out)

    dwt_kernel<<<(CATOT + 255)/256, 256, 0, stream>>>(f, cA, cD);
    im2col_kernel<<<(BATCH*CALEN*768 + 255)/256, 256, 0, stream>>>(cA, colb);
    gemm(stream, 0, colb, 768, WVp(i,0), WVp(i,1), caf, 256, nullptr, BATCH*CALEN, 768, 256);
    im2col_kernel<<<(BATCH*CALEN*768 + 255)/256, 256, 0, stream>>>(cD, colb);
    gemm(stream, 0, colb, 768, WVp(i,2), WVp(i,3), cdf, 256, nullptr, BATCH*CALEN, 768, 256);
    recon_kernel<<<(int)((R*256 + 255)/256), 256, 0, stream>>>(caf, cdf, rect);
    ln_kernel<<<(int)R, 256, 0, stream>>>(rect, f, WVp(i,4), WVp(i,5), f);                  // LN(x+rec)
  }

  // ---------- gated fusion + projection ----------
  float* gb = xz;
  gemm(stream, 3, t, 256, gate_w, gate_b, gb, 256, nullptr, (int)R, 256, 256);              // sigmoid gate
  combine_kernel<<<(int)((R*256 + 255)/256), 256, 0, stream>>>(t, f, gb, hbuf);
  proj_kernel<<<(BATCH*PREDL + 255)/256, 256, 0, stream>>>(hbuf, proj_w, proj_b, (float*)d_out);
}
